// GraphAttentionLayer_7198365188484
// MI455X (gfx1250) — matrix operand order repaired and checked by reading, not compiled, not run on hardware
//
#include <hip/hip_runtime.h>

// ---------------------------------------------------------------------------
// GraphAttentionLayer for MI455X (gfx1250, wave32, WMMA).
//
// Key simplification: softmax over a row whose live entries all equal z[i]
// reduces to uniform 1/deg(i) over neighbors (dead entries underflow to 0).
// So h_prime = D^-1 * A * h, computed with v_wmma_f32_16x16x32_f16 where the
// 0/1 adjacency is converted to f16 on the fly (exact) and h is f16 with f32
// accumulation. Adjacency (268 MB) is streamed non-temporally (read once);
// h_f16^T (2 MB) stays L2-resident.
// ---------------------------------------------------------------------------

typedef _Float16 half_t;
typedef __attribute__((ext_vector_type(16))) _Float16 v16h;
typedef __attribute__((ext_vector_type(8)))  _Float16 v8h;
typedef __attribute__((ext_vector_type(8)))  float    v8f;
typedef __attribute__((ext_vector_type(4)))  int      v4i;

#define N_NODES 8192
#define F_IN    512
#define F_OUT   128
#define H_DIM   16

static __device__ __forceinline__ v16h join16(v8h lo, v8h hi) {
  return __builtin_shufflevector(lo, hi, 0,1,2,3,4,5,6,7,8,9,10,11,12,13,14,15);
}

// ---- Kernel 0: W [512,128] f32 -> wT [128,512] f16 (contiguous B frags) ----
__global__ void wt_convert(const float* __restrict__ W, half_t* __restrict__ wT) {
  int idx = blockIdx.x * 256 + threadIdx.x;        // 65536 elements
  int n = idx >> 9;                                 // 0..127
  int k = idx & 511;                                // 0..511
  wT[idx] = (half_t)W[k * F_OUT + n];
}

// ---- Kernel 1: h = x @ W via WMMA f16. Writes h (f32) and hT (f16, N-major) -
__global__ __launch_bounds__(256) void gemm_xw(
    const float* __restrict__ x, const half_t* __restrict__ wT,
    float* __restrict__ h, half_t* __restrict__ hT)
{
  __shared__ half_t a_sh[2][16 * 40];              // padded stride: 40 halves
  const int t    = threadIdx.x;
  const int wv   = t >> 5;
  const int lane = t & 31;
  const int m0   = blockIdx.x * 16;
  const int nloc = lane & 15;
  const int n    = wv * 16 + nloc;
  const int khalf = (lane < 16) ? 0 : 8;           // unused; A frag also uses kbB (K0-15 / K16-31)
  const int kbB   = (lane < 16) ? 0 : 16;          // B frag: K0-15 / K16-31

  const int sr = t >> 4;                           // 16 rows, 16 thr/row
  const int sk = (t & 15) * 2;                     // 2 f32 each

  v8f acc = {};

  // prologue stage kk=0
  {
    const float* xp = x + (size_t)(m0 + sr) * F_IN + sk;
    half_t* p = &a_sh[0][sr * 40 + sk];
    p[0] = (half_t)xp[0]; p[1] = (half_t)xp[1];
  }
  __syncthreads();

  for (int step = 0; step < F_IN / 32; ++step) {
    const int kk = step * 32;
    if (kk + 32 < F_IN) {
      const float* xp = x + (size_t)(m0 + sr) * F_IN + kk + 32 + sk;
      half_t* p = &a_sh[(step + 1) & 1][sr * 40 + sk];
      p[0] = (half_t)xp[0]; p[1] = (half_t)xp[1];
    }
    const half_t* ab = &a_sh[step & 1][nloc * 40 + kbB];
    v16h afrag = join16(*(const v8h*)ab, *(const v8h*)(ab + 8));
    const half_t* bb = wT + (size_t)n * F_IN + kk + kbB;
    v16h bfrag = join16(*(const v8h*)bb, *(const v8h*)(bb + 8));
    acc = __builtin_amdgcn_wmma_f32_16x16x32_f16(false, afrag, false, bfrag,
                                                 (short)0, acc, false, false);
    __syncthreads();
  }

  // C layout: VGPR v -> M=v (lanes 0-15) / M=v+8 (lanes 16-31), N=lane&15
  #pragma unroll
  for (int v = 0; v < 8; ++v) {
    int m = v + ((lane < 16) ? 0 : 8);
    float val = acc[v];
    h[(size_t)(m0 + m) * F_OUT + n] = val;
    hT[(size_t)n * N_NODES + m0 + m] = (half_t)val;
  }
}

// ---- Kernel 2: per-node MLP -> qmp partial per block (deterministic) -------
__global__ __launch_bounds__(256) void attn_mlp(
    const float* __restrict__ h,
    const float* __restrict__ w1,  const float* __restrict__ b1,
    const float* __restrict__ w21, const float* __restrict__ b21,
    const float* __restrict__ w22, const float* __restrict__ b22,
    float* __restrict__ qpart)
{
  __shared__ float we1[H_DIM][F_OUT];   // folded concat weight: w1[:, :128]+w1[:,128:]
  __shared__ float red[256];
  const int t = threadIdx.x;
  for (int i = t; i < H_DIM * F_OUT; i += 256) {
    int k = i >> 7, f = i & 127;
    we1[k][f] = w1[k * (2 * F_OUT) + f] + w1[k * (2 * F_OUT) + F_OUT + f];
  }
  __syncthreads();

  const int node = blockIdx.x * 256 + t;
  float u[H_DIM];
  #pragma unroll
  for (int k = 0; k < H_DIM; ++k) u[k] = 0.f;
  const float* hr = h + (size_t)node * F_OUT;
  for (int f = 0; f < F_OUT; ++f) {
    float hv = hr[f];
    #pragma unroll
    for (int k = 0; k < H_DIM; ++k) u[k] = fmaf(hv, we1[k][f], u[k]);
  }
  float mu = b21[0], lv = b22[0];
  #pragma unroll
  for (int k = 0; k < H_DIM; ++k) {
    float uk = fmaxf(u[k] + b1[k], 0.f);
    mu = fmaf(uk, w21[k], mu);
    lv = fmaf(uk, w22[k], lv);
  }
  float sigma = (lv > 20.f) ? lv : log1pf(__expf(lv));     // softplus
  red[t] = 0.5f * mu * mu - logf(sigma);                   // qmp contribution
  __syncthreads();
  for (int s = 128; s > 0; s >>= 1) {
    if (t < s) red[t] += red[t + s];
    __syncthreads();
  }
  if (t == 0) qpart[blockIdx.x] = red[0];
}

__global__ void qmp_final(const float* __restrict__ qpart, float* __restrict__ out) {
  if (threadIdx.x == 0) {
    float s = 0.f;
    for (int i = 0; i < N_NODES / 256; ++i) s += qpart[i];  // fixed order
    out[0] = s;
  }
}

// ---- Kernel 3: out = elu(D^-1 A h) via WMMA over streamed adjacency --------
// Block: 256 threads = 8 waves. M=64 rows, N=128 cols, K loop over 8192.
__global__ __launch_bounds__(256) void attn_aggregate(
    const int* __restrict__ adj, const half_t* __restrict__ hT,
    float* __restrict__ out)
{
  __shared__ half_t a_sh[2][64 * 40];              // 64 rows x 32 K, stride 40
  __shared__ int    degsh[64][4];
  __shared__ float  deginv[64];

  const int t    = threadIdx.x;
  const int wv   = t >> 5;
  const int lane = t & 31;
  const int r0   = blockIdx.x * 64;
  const int nloc = lane & 15;
  const int n    = wv * 16 + nloc;
  const int khalf = (lane < 16) ? 0 : 8;
  const int kbB   = (lane < 16) ? 0 : 16;

  const int sr = t >> 2;                           // 64 rows, 4 thr/row
  const int sk = (t & 3) * 8;                      // 8 ints each
  int degp = 0;

  auto stage = [&](int buf, int kk) {
    const v4i* gp = (const v4i*)(adj + (size_t)(r0 + sr) * N_NODES + kk + sk);
    v4i a0 = __builtin_nontemporal_load(gp);       // adj is read exactly once
    v4i a1 = __builtin_nontemporal_load(gp + 1);
    v8h hv;
    #pragma unroll
    for (int j = 0; j < 4; ++j) { hv[j]     = (half_t)a0[j]; degp += a0[j]; }
    #pragma unroll
    for (int j = 0; j < 4; ++j) { hv[4 + j] = (half_t)a1[j]; degp += a1[j]; }
    *(v8h*)&a_sh[buf][sr * 40 + sk] = hv;          // one ds_store_b128
  };

  v8f acc[4] = {};
  stage(0, 0);
  __syncthreads();

  for (int step = 0; step < N_NODES / 32; ++step) {
    const int kk = step * 32;
    if (kk + 32 < N_NODES) stage((step + 1) & 1, kk + 32);

    // B frag: 16 consecutive K halves of hT row n (L2-resident), two b128s
    const half_t* bb = hT + (size_t)n * N_NODES + kk + kbB;
    v16h bfrag = join16(*(const v8h*)bb, *(const v8h*)(bb + 8));

    #pragma unroll
    for (int mt = 0; mt < 4; ++mt) {
      const half_t* ab = &a_sh[step & 1][(mt * 16 + nloc) * 40 + kbB];
      v16h afrag = join16(*(const v8h*)ab, *(const v8h*)(ab + 8));
      acc[mt] = __builtin_amdgcn_wmma_f32_16x16x32_f16(false, afrag, false, bfrag,
                                                       (short)0, acc[mt], false, false);
    }
    __syncthreads();
  }

  // per-row degree: reduce 4 partials per row
  degsh[sr][t & 3] = degp;
  __syncthreads();
  if (t < 64) {
    int d = degsh[t][0] + degsh[t][1] + degsh[t][2] + degsh[t][3];
    deginv[t] = (d > 0) ? (1.0f / (float)d) : 0.f;   // deg==0 -> row of zeros -> elu(0)=0
  }
  __syncthreads();

  #pragma unroll
  for (int mt = 0; mt < 4; ++mt) {
    #pragma unroll
    for (int v = 0; v < 8; ++v) {
      int m = mt * 16 + v + ((lane < 16) ? 0 : 8);
      float val = acc[mt][v] * deginv[m];
      val = (val > 0.f) ? val : (__expf(val) - 1.0f);  // elu
      out[(size_t)(r0 + m) * F_OUT + n] = val;
    }
  }
}

// ---------------------------------------------------------------------------
extern "C" void kernel_launch(void* const* d_in, const int* in_sizes, int n_in,
                              void* d_out, int out_size, void* d_ws, size_t ws_size,
                              hipStream_t stream) {
  (void)in_sizes; (void)n_in; (void)out_size; (void)ws_size;
  const float* x   = (const float*)d_in[0];
  const int*   adj = (const int*)  d_in[1];
  const float* W   = (const float*)d_in[2];
  const float* w1  = (const float*)d_in[3];
  const float* b1  = (const float*)d_in[4];
  const float* w21 = (const float*)d_in[5];
  const float* b21 = (const float*)d_in[6];
  const float* w22 = (const float*)d_in[7];
  const float* b22 = (const float*)d_in[8];
  float* out = (float*)d_out;

  char* ws = (char*)d_ws;
  half_t* wT = (half_t*)ws;                                  // 128 KB
  float*  h  = (float*) (ws + (128 << 10));                  // 4 MB
  half_t* hT = (half_t*)(ws + (128 << 10) + (4 << 20));      // 2 MB
  float*  qp = (float*) (ws + (128 << 10) + (6 << 20));      // 128 B

  wt_convert    <<<256, 256, 0, stream>>>(W, wT);
  gemm_xw       <<<N_NODES / 16, 256, 0, stream>>>(x, wT, h, hT);
  attn_mlp      <<<N_NODES / 256, 256, 0, stream>>>(h, w1, b1, w21, b21, w22, b22, qp);
  qmp_final     <<<1, 32, 0, stream>>>(qp, out + (size_t)N_NODES * F_OUT);
  attn_aggregate<<<N_NODES / 64, 256, 0, stream>>>(adj, hT, out);
}
